// LightweightConv1d_71090298683650
// MI455X (gfx1250) — matrix operand order repaired and checked by reading, not compiled, not run on hardware
//
#include <hip/hip_runtime.h>
#include <hip/hip_bf16.h>
#include <math.h>
#include <stdint.h>

// ---------- types ----------
typedef __attribute__((ext_vector_type(16))) __bf16 v16bf;
typedef __attribute__((ext_vector_type(8)))  float  v8f;
typedef __attribute__((ext_vector_type(4)))  unsigned int u32x4;

#define H_DIM 1024
#define NH 16
#define KS 31
#define BQ 4
#define SQ 4096
#define MROWS (BQ * SQ)   // 16384
#define EPS 1e-5f

// ===================== softmax + causal mask of conv weights =====================
__global__ void softmax_convw_kernel(const float* __restrict__ conv_w,
                                     float* __restrict__ sw) {
    int h = threadIdx.x;
    if (h < NH) {
        float m = -1e30f;
        #pragma unroll
        for (int j = 0; j < KS; ++j) m = fmaxf(m, conv_w[h * KS + j]);
        float e[KS];
        float s = 0.f;
        #pragma unroll
        for (int j = 0; j < KS; ++j) { e[j] = __expf(conv_w[h * KS + j] - m); s += e[j]; }
        float inv = 1.0f / s;
        #pragma unroll
        for (int j = 0; j < KS; ++j)
            sw[h * KS + j] = (j < (KS / 2 + 1)) ? e[j] * inv : 0.0f;  // causal mask (pivot=16)
    }
}

// ===================== fp32 -> bf16 weight conversion (row-major [N][K]) =====================
__global__ void cvt_weights_kernel(const float* __restrict__ w_in,
                                   const float* __restrict__ w_out,
                                   __bf16* __restrict__ wb_in,
                                   __bf16* __restrict__ wb_out) {
    size_t i = (size_t)blockIdx.x * blockDim.x + threadIdx.x;
    if (i < (size_t)H_DIM * H_DIM) {
        wb_in[i]  = (__bf16)w_in[i];
        wb_out[i] = (__bf16)w_out[i];
    }
}

// ===================== fp32 -> bf16 activation conversion (4 elems/thread) =====================
__global__ __launch_bounds__(256)
void cvt_act_kernel(const float* __restrict__ x, __bf16* __restrict__ y) {
    size_t i = (size_t)blockIdx.x * blockDim.x + threadIdx.x;   // over n/4 quads
    const float4 v = ((const float4*)x)[i];
    union { __bf16 h[4]; uint2 u; } t;
    t.h[0] = (__bf16)v.x; t.h[1] = (__bf16)v.y;
    t.h[2] = (__bf16)v.z; t.h[3] = (__bf16)v.w;
    ((uint2*)y)[i] = t.u;
}

// ===================== CDNA5 async global->LDS staging =====================
__device__ __forceinline__ void async_ld_b128(unsigned lds_addr, const void* gaddr) {
    asm volatile("global_load_async_to_lds_b128 %0, %1, off"
                 :: "v"(lds_addr), "v"(gaddr) : "memory");
}
__device__ __forceinline__ void wait_async0() {
    asm volatile("s_wait_asynccnt 0" ::: "memory");
}

// Stage nchunks 16B chunks (8 bf16 each); tile rows of 32 k, padded LDS stride 40.
__device__ __forceinline__ void stage_async(const __bf16* __restrict__ g, int gstride,
                                            __bf16* l, int nchunks, int tid) {
    #pragma unroll
    for (int i = tid; i < nchunks; i += 256) {
        int rl = i >> 2, kc = (i & 3) * 8;
        const __bf16* src = g + (size_t)rl * gstride + kc;
        unsigned dst = (unsigned)(uintptr_t)(l + rl * 40 + kc);   // flat addr low 32 = LDS offset
        async_ld_b128(dst, src);
    }
}

// ===================== WMMA fragment loaders (ISA 7.12.2 layouts) =====================
// A (16x32 bf16): lane l<16 -> M=l, K in [0..7]+[16..23]; lane>=16 -> M=l-16, K in [8..15]+[24..31]
__device__ __forceinline__ v16bf load_fragA(const __bf16* tile, int stride, int lane) {
    int r = lane & 15;
    int c = (lane >> 4) * 8;
    const __bf16* p = tile + r * stride + c;
    union { v16bf v; u32x4 u[2]; } f;
    f.u[0] = *(const u32x4*)(p);
    f.u[1] = *(const u32x4*)(p + 16);
    return f.v;
}
// B (32x16 bf16) fed from n-major LDS tile Bs[n][k]: lane l -> N=l&15, K in (l>>4)*8+[0..7] + [16..23]
__device__ __forceinline__ v16bf load_fragB(const __bf16* tile, int stride, int lane) {
    int n  = lane & 15;
    int kb = (lane >> 4) * 8;
    const __bf16* p = tile + n * stride + kb;
    union { v16bf v; u32x4 u[2]; } f;
    f.u[0] = *(const u32x4*)(p);
    f.u[1] = *(const u32x4*)(p + 16);
    return f.v;
}

// ===================== bf16 WMMA GEMM: C[M,N] = A[M,K] * W[N,K]^T + bias =====================
// Block: 256 threads = 8 wave32s; block tile 128(M) x 256(N); wave tile 64x64; K-step 32.
// Double-buffered LDS, async global->LDS copy overlapped with WMMA compute.
__global__ __launch_bounds__(256)
void gemm_wmma_kernel(const __bf16* __restrict__ A,     // [M][K] bf16
                      const __bf16* __restrict__ Wb,    // [N][K] bf16 (row-major == WMMA-B feed order)
                      const float* __restrict__ bias,   // [N]
                      float* __restrict__ C,            // [M][N]
                      int M, int N, int K) {
    constexpr int BM = 128, BN = 256, BK = 32, AST = 40;   // padded LDS row stride (bf16)
    __shared__ __align__(16) __bf16 As[2][BM * AST];
    __shared__ __align__(16) __bf16 Bs[2][BN * AST];

    const int tid  = threadIdx.x;
    const int wave = tid >> 5;
    const int lane = tid & 31;
    const int wm = (wave >> 2) * 64;
    const int wn = (wave & 3) * 64;
    const int blockM = blockIdx.y * BM;
    const int blockN = blockIdx.x * BN;

    const __bf16* Ag = A  + (size_t)blockM * K;
    const __bf16* Bg = Wb + (size_t)blockN * K;

    v8f acc[4][4] = {};

    // prologue: stage k0 = 0 into buffer 0
    stage_async(Ag, K, &As[0][0], BM * 4, tid);
    stage_async(Bg, K, &Bs[0][0], BN * 4, tid);

    int cur = 0;
    for (int k0 = 0; k0 < K; k0 += BK) {
        wait_async0();
        __syncthreads();                       // buf(cur) fully resident for all waves

        int nxt = cur ^ 1;
        if (k0 + BK < K) {                     // stream next tile while computing this one
            stage_async(Ag + k0 + BK, K, &As[nxt][0], BM * 4, tid);
            stage_async(Bg + k0 + BK, K, &Bs[nxt][0], BN * 4, tid);
        }

        v16bf afrag[4];
        #pragma unroll
        for (int mi = 0; mi < 4; ++mi)
            afrag[mi] = load_fragA(&As[cur][(wm + mi * 16) * AST], AST, lane);
        #pragma unroll
        for (int ni = 0; ni < 4; ++ni) {
            v16bf bfrag = load_fragB(&Bs[cur][(wn + ni * 16) * AST], AST, lane);
            #pragma unroll
            for (int mi = 0; mi < 4; ++mi)
                acc[mi][ni] = __builtin_amdgcn_wmma_f32_16x16x32_bf16(
                    false, afrag[mi], false, bfrag, (short)0, acc[mi][ni], false, false);
        }
        cur = nxt;
    }

    // ---- store: C/D layout: lanes 0-15 -> M=v (N=lane), lanes 16-31 -> M=8+v ----
    const int cn  = lane & 15;
    const int cmh = (lane >> 4) * 8;
    #pragma unroll
    for (int ni = 0; ni < 4; ++ni) {
        int col = blockN + wn + ni * 16 + cn;
        float bv = bias[col];
        #pragma unroll
        for (int mi = 0; mi < 4; ++mi) {
            int row0 = blockM + wm + mi * 16 + cmh;
            #pragma unroll
            for (int v = 0; v < 8; ++v)
                C[(size_t)(row0 + v) * N + col] = acc[mi][ni][v] + bv;
        }
    }
}

// ===================== causal depthwise conv (16 live taps), fp32 in -> bf16 out =====================
__global__ __launch_bounds__(256)
void conv_kernel(const float* __restrict__ x1, const float* __restrict__ sw,
                 __bf16* __restrict__ out) {
    const int b  = blockIdx.x >> 6;           // 64 s-tiles per batch
    const int s0 = (blockIdx.x & 63) * 64;
    const int h0 = blockIdx.y * 64;
    const int head = blockIdx.y;              // H/NH == 64 == tile width

    __shared__ float xs[79 * 64];             // rows: s0-15 .. s0+63
    __shared__ float wsm[16];
    const int tid = threadIdx.x;
    if (tid < 16) wsm[tid] = sw[head * KS + tid];

    const float* xb = x1 + (size_t)b * SQ * H_DIM;
    for (int i = tid; i < 79 * 64; i += 256) {
        int sr = i >> 6, h = i & 63;
        int s = s0 + sr - 15;
        xs[i] = (s >= 0) ? xb[(size_t)s * H_DIM + h0 + h] : 0.0f;
    }
    __syncthreads();

    const int h   = tid & 63;
    const int sl0 = tid >> 6;
    #pragma unroll
    for (int i = 0; i < 16; ++i) {
        int sl = sl0 + i * 4;
        float acc = 0.0f;
        #pragma unroll
        for (int j = 0; j < 16; ++j) acc = fmaf(wsm[j], xs[(sl + j) * 64 + h], acc);
        out[((size_t)(b * SQ + s0 + sl)) * H_DIM + h0 + h] = (__bf16)acc;
    }
}

// ===================== residual + layernorm =====================
__global__ __launch_bounds__(256)
void layernorm_kernel(const float* __restrict__ hs, const float* __restrict__ x2,
                      const float* __restrict__ gamma, const float* __restrict__ beta,
                      float* __restrict__ out) {
    const int r = blockIdx.x;
    const float* a = hs + (size_t)r * H_DIM;
    const float* b = x2 + (size_t)r * H_DIM;
    const int tid = threadIdx.x;

    float y[4], s = 0.f, s2 = 0.f;
    #pragma unroll
    for (int i = 0; i < 4; ++i) {
        int c = tid + i * 256;
        y[i] = a[c] + b[c];
        s += y[i];
        s2 += y[i] * y[i];
    }
    __shared__ float rs[256], rq[256];
    rs[tid] = s; rq[tid] = s2;
    __syncthreads();
    for (int off = 128; off > 0; off >>= 1) {
        if (tid < off) { rs[tid] += rs[tid + off]; rq[tid] += rq[tid + off]; }
        __syncthreads();
    }
    float mu  = rs[0] * (1.0f / H_DIM);
    float var = rq[0] * (1.0f / H_DIM) - mu * mu;
    float inv = rsqrtf(var + EPS);
    #pragma unroll
    for (int i = 0; i < 4; ++i) {
        int c = tid + i * 256;
        out[(size_t)r * H_DIM + c] = (y[i] - mu) * inv * gamma[c] + beta[c];
    }
}

// ===================== launch =====================
extern "C" void kernel_launch(void* const* d_in, const int* in_sizes, int n_in,
                              void* d_out, int out_size, void* d_ws, size_t ws_size,
                              hipStream_t stream) {
    const float* hs     = (const float*)d_in[0];
    const float* W_in   = (const float*)d_in[1];
    const float* b_in   = (const float*)d_in[2];
    const float* conv_w = (const float*)d_in[3];
    const float* W_out  = (const float*)d_in[4];
    const float* b_out  = (const float*)d_in[5];
    const float* gamma  = (const float*)d_in[6];
    const float* beta   = (const float*)d_in[7];
    float* out = (float*)d_out;

    char* ws = (char*)d_ws;
    float*  x1    = (float*)ws;                          // 64 MiB : GEMM1 out, reused for GEMM2 out
    __bf16* hsb   = (__bf16*)(ws + (64ull << 20));       // 32 MiB : hs in bf16 (GEMM1 A) ...
    __bf16* convb = (__bf16*)(ws + (64ull << 20));       //          ... reused as conv out (GEMM2 A)
    __bf16* wbIn  = (__bf16*)(ws + (96ull << 20));       //  2 MiB
    __bf16* wbOut = (__bf16*)(ws + (98ull << 20));       //  2 MiB
    float*  sw    = (float*)(ws + (100ull << 20));       // 16*31 floats

    softmax_convw_kernel<<<1, 32, 0, stream>>>(conv_w, sw);
    cvt_weights_kernel<<<(H_DIM * H_DIM) / 256, 256, 0, stream>>>(W_in, W_out, wbIn, wbOut);
    cvt_act_kernel<<<(MROWS * H_DIM) / (256 * 4), 256, 0, stream>>>(hs, hsb);

    dim3 ggrid(H_DIM / 256, MROWS / 128);  // (4, 128)
    gemm_wmma_kernel<<<ggrid, 256, 0, stream>>>(hsb, wbIn, b_in, x1, MROWS, H_DIM, H_DIM);

    conv_kernel<<<dim3((BQ * SQ) / 64, H_DIM / 64), 256, 0, stream>>>(x1, sw, convb);

    gemm_wmma_kernel<<<ggrid, 256, 0, stream>>>(convb, wbOut, b_out, x1, MROWS, H_DIM, H_DIM);

    layernorm_kernel<<<MROWS, 256, 0, stream>>>(hs, x1, gamma, beta, out);
}
